// DimerInteractionEnergyModel_76897094467607
// MI455X (gfx1250) — matrix operand order repaired and checked by reading, not compiled, not run on hardware
//
#include <hip/hip_runtime.h>
#include <cmath>

// ---------------------------------------------------------------------------
// DimerInteractionEnergyModel on MI455X (gfx1250, wave32, WMMA)
//
// TP per edge: out[e,k] = sum_{i<22,j<32} ef[e,i]*y[idx[e],j]*w[i,j,k]
// == GEMM [16 edges x 704] x [704 x 32] per wave tile, K-blocked by 32 where
// K-block i is ef[e,i] * y_row (rank-1 structure) -> rescale one bf16
// A-fragment per scalar channel, 2x v_wmma_f32_16x16x32_bf16 per channel.
// f32 accumulate, SiLU, atomic scatter (segment_sum).
//
// v3: A-fragment rescale done in native packed bf16 (v_pk_mul_bf16 via
// inline asm) against a once-per-tile bf16 y-fragment: 9 VALU/channel
// instead of 16, cutting VALU pressure that co-executes with the XDL WMMAs.
// ---------------------------------------------------------------------------

#define N0C 50000
#define N1C 50000
#define EC  400000
#define DIMC 32
#define NSCAL 22
#define NLAYERS 3
#define EFSTRIDE 24                       // ef padded 22 -> 24 floats/edge
#define FRAG_HALVES (NSCAL * 2 * 32 * 16) // 22528 bf16 per (layer,dir) slice
#define FRAG_BYTES  (FRAG_HALVES * 2)     // 45056 B

typedef __attribute__((ext_vector_type(16))) __bf16 v16bf;
typedef __attribute__((ext_vector_type(8)))  float  v8f;

__device__ __forceinline__ float silu_f(float x) {
    return x / (1.0f + __expf(-x));
}

// ---- atomic embedding: y[n,c] = emb_w[z[n],c] + emb_b[c] (y0 | y1 stacked) --
__global__ void k_embed(const int* __restrict__ z0, const int* __restrict__ z1,
                        const float* __restrict__ emb_w, const float* __restrict__ emb_b,
                        float* __restrict__ y) {
    int t = blockIdx.x * blockDim.x + threadIdx.x;
    if (t >= (N0C + N1C) * DIMC) return;
    int n = t >> 5, c = t & 31;
    int z = (n < N0C) ? z0[n] : z1[n - N0C];
    y[t] = emb_w[z * DIMC + c] + emb_b[c];
}

// ---- edge scalar features: 21 gaussians + 1.0, pre-scaled by TP+N norm -----
__global__ void k_edgefeat(const float* __restrict__ r, float* __restrict__ ef,
                           float scale) {
    int e = blockIdx.x * blockDim.x + threadIdx.x;
    if (e >= EC) return;
    float rv = r[e];
    float* o = ef + (size_t)e * EFSTRIDE;
#pragma unroll
    for (int i = 0; i < 21; ++i) {
        float d = (rv - 0.25f * (float)i) * 4.0f;  // (r-mu)/STEP, STEP=0.25
        o[i] = __expf(-0.5f * d * d) * scale;
    }
    o[21] = scale;     // sh0 channel (Y0 == 1)
    o[22] = 0.0f;
    o[23] = 0.0f;
}

// ---- prepack w[l][i][j][k] f32 -> bf16 B-fragments in ISA lane layout ------
// B (32x16, K x N): lane(0..15) holds col N=lane, K={0..7,16..23} in 16 halves;
// lane+16 holds col N=lane, K={8..15,24..31}.  Two N-tiles cover k=0..31.
__global__ void k_prepack(const float* __restrict__ w_s2d,
                          const float* __restrict__ w_d2s,
                          __bf16* __restrict__ frag) {
    int t = blockIdx.x * blockDim.x + threadIdx.x;
    if (t >= NLAYERS * 2 * NSCAL * 2 * 32) return;
    int lane = t & 31;
    int nt   = (t >> 5) & 1;
    int i    = (t >> 6) % NSCAL;
    int ld   = t / (64 * NSCAL);           // l*2 + dir
    int dir  = ld & 1, l = ld >> 1;
    const float* w  = (dir == 0) ? w_s2d : w_d2s;
    const float* wi = w + (((size_t)l * NSCAL + i) * 32) * 32;   // [j][k]
    __bf16* out = frag + ((((size_t)ld * NSCAL + i) * 2 + nt) * 32 + lane) * 16;
    int kbase = (lane >> 4) * 8;           // K (== j) base for this lane half
    int n     = nt * 16 + (lane & 15);     // output channel k
#pragma unroll
    for (int h = 0; h < 16; ++h)
        out[h] = (__bf16)wi[(kbase + h + ((h >> 3) << 3)) * 32 + n];
}

// ---- main TP GEMM: one wave = 16 edges, full K=704 via 22 bf16 WMMA pairs --
__launch_bounds__(256)
__global__ void k_tp(const float* __restrict__ ef, const float* __restrict__ y_in,
                     const int* __restrict__ idxA, const int* __restrict__ idxB,
                     const __bf16* __restrict__ wfrag, float* __restrict__ d_acc) {
    __shared__ __align__(32) __bf16 sB[FRAG_HALVES];   // 44 KB weight stage
    {   // cooperative stage: whole (layer,dir) weight slice into LDS
        const float4* src = (const float4*)wfrag;
        float4* dst = (float4*)sB;
        for (int idx = threadIdx.x; idx < FRAG_BYTES / 16; idx += 256)
            dst[idx] = src[idx];
    }
    __syncthreads();

    int wave = threadIdx.x >> 5;
    int lane = threadIdx.x & 31;
    int base = (blockIdx.x * 8 + wave) * 16;   // 400000 = 16*25000, exact
    int m    = lane & 15;
    int e    = base + m;
    int rowA = idxA[e];
    int rowB = idxB[e];

    // A-fragment of y (16x32 bf16 layout): lane<16 holds K={klo..klo+7,klo+16..klo+23}
    int klo = (lane < 16) ? 0 : 8;
    const float* yr = y_in + (size_t)rowA * 32;
    float yf[16];
    {
        float4 a0 = *(const float4*)(yr + klo);
        float4 a1 = *(const float4*)(yr + klo + 4);
        float4 a2 = *(const float4*)(yr + klo + 16);
        float4 a3 = *(const float4*)(yr + klo + 20);
        yf[0]=a0.x; yf[1]=a0.y; yf[2]=a0.z; yf[3]=a0.w;
        yf[4]=a1.x; yf[5]=a1.y; yf[6]=a1.z; yf[7]=a1.w;
        yf[8]=a2.x; yf[9]=a2.y; yf[10]=a2.z; yf[11]=a2.w;
        yf[12]=a3.x; yf[13]=a3.y; yf[14]=a3.z; yf[15]=a3.w;
    }

    // once-per-tile: pack y fragment to bf16 pairs (8 dwords)
    unsigned ybu[8];
#pragma unroll
    for (int p = 0; p < 8; ++p)
        asm("v_cvt_pk_bf16_f32 %0, %1, %2"
            : "=v"(ybu[p]) : "v"(yf[2 * p]), "v"(yf[2 * p + 1]));

    // per-edge scalar channels (already norm-scaled)
    float s[24];
    {
        const float4* efp = (const float4*)(ef + (size_t)e * EFSTRIDE);
#pragma unroll
        for (int q = 0; q < 6; ++q) {
            float4 v = efp[q];
            s[q*4+0]=v.x; s[q*4+1]=v.y; s[q*4+2]=v.z; s[q*4+3]=v.w;
        }
    }

    v8f acc0 = {}; v8f acc1 = {};
#pragma unroll
    for (int i = 0; i < NSCAL; ++i) {
        // broadcast bf16 pair of the channel scalar (1 instruction)
        unsigned sb;
        asm("v_cvt_pk_bf16_f32 %0, %1, %1" : "=v"(sb) : "v"(s[i]));
        // A_i = s_i * Y in packed bf16 (8x v_pk_mul_bf16)
        unsigned au[8];
#pragma unroll
        for (int p = 0; p < 8; ++p)
            asm("v_pk_mul_bf16 %0, %1, %2" : "=v"(au[p]) : "v"(sb), "v"(ybu[p]));
        v16bf a;
        __builtin_memcpy(&a, au, sizeof(a));
        v16bf b0 = *(const v16bf*)(sB + (i * 2 + 0) * (32 * 16) + lane * 16);
        v16bf b1 = *(const v16bf*)(sB + (i * 2 + 1) * (32 * 16) + lane * 16);
        acc0 = __builtin_amdgcn_wmma_f32_16x16x32_bf16(false, a, false, b0,
                                                       (short)0, acc0, false, false);
        acc1 = __builtin_amdgcn_wmma_f32_16x16x32_bf16(false, a, false, b1,
                                                       (short)0, acc1, false, false);
    }

    // C layout: vgpr v, lane L -> M = v + 8*(L>=16), N = (L&15) (+16 for tile 1)
#pragma unroll
    for (int v = 0; v < 8; ++v) {
        int srcLane = v + ((lane >> 4) << 3);
        int ridx = __shfl(rowB, srcLane, 32);        // ds_bpermute
        float x0 = silu_f(acc0[v]);
        float x1 = silu_f(acc1[v]);
        atomicAdd(d_acc + (size_t)ridx * 32 + (lane & 15), x0);
        atomicAdd(d_acc + (size_t)ridx * 32 + 16 + (lane & 15), x1);
    }
}

// ---- helpers ---------------------------------------------------------------
__global__ void k_zero(float* __restrict__ p, int n) {
    int t = blockIdx.x * blockDim.x + threadIdx.x;
    if (t < n) p[t] = 0.0f;
}
__global__ void k_addzero(float* __restrict__ y, float* __restrict__ d, int n) {
    int t = blockIdx.x * blockDim.x + threadIdx.x;
    if (t < n) { y[t] += d[t]; d[t] = 0.0f; }
}
__global__ void k_out_init(float* out) { out[0] = 0.0f; }

// ---- readout: sum(silu(y @ ro_w + ro_b)) with wave32 reduction -------------
__global__ void k_readout(const float* __restrict__ y, const float* __restrict__ ro_w,
                          const float* __restrict__ ro_b, float* __restrict__ out) {
    int t = blockIdx.x * blockDim.x + threadIdx.x;
    float v = 0.0f;
    if (t < N0C + N1C) {
        const float* row = y + (size_t)t * 32;
        float acc = ro_b[0];
#pragma unroll
        for (int c = 0; c < 32; ++c) acc += row[c] * ro_w[c];
        v = silu_f(acc);
    }
#pragma unroll
    for (int off = 16; off > 0; off >>= 1) v += __shfl_down(v, off, 32);
    if ((threadIdx.x & 31) == 0) atomicAdd(out, v);
}

extern "C" void kernel_launch(void* const* d_in, const int* in_sizes, int n_in,
                              void* d_out, int out_size, void* d_ws, size_t ws_size,
                              hipStream_t stream) {
    const int*   z0    = (const int*)d_in[0];
    const int*   z1    = (const int*)d_in[1];
    const int*   src   = (const int*)d_in[2];
    const int*   dst   = (const int*)d_in[3];
    const float* r     = (const float*)d_in[4];
    // d_in[5] = r_hat: l=1 SH has no 0e coupling path -> unused
    const float* emb_w = (const float*)d_in[6];
    const float* emb_b = (const float*)d_in[7];
    const float* w_s2d = (const float*)d_in[8];
    const float* w_d2s = (const float*)d_in[9];
    const float* ro_w  = (const float*)d_in[10];
    const float* ro_b  = (const float*)d_in[11];
    float* out = (float*)d_out;

    // workspace layout (floats): y[100000*32] | d[100000*32] | ef[E*24] | wfrag(bf16)
    const size_t NY = (size_t)(N0C + N1C) * DIMC;     // 3.2M
    float*  y     = (float*)d_ws;
    float*  dacc  = y + NY;
    float*  ef    = dacc + NY;
    __bf16* wfrag = (__bf16*)(ef + (size_t)EC * EFSTRIDE);

    const float scale = 1.0f / sqrtf((float)(NSCAL * DIMC) * (float)(N0C + N1C));

    k_embed<<<(int)((NY + 255) / 256), 256, 0, stream>>>(z0, z1, emb_w, emb_b, y);
    k_edgefeat<<<(EC + 255) / 256, 256, 0, stream>>>(r, ef, scale);
    k_prepack<<<(NLAYERS * 2 * NSCAL * 2 * 32 + 255) / 256, 256, 0, stream>>>(
        w_s2d, w_d2s, wfrag);
    k_zero<<<(int)((NY + 255) / 256), 256, 0, stream>>>(dacc, (int)NY);

    const int tpBlocks = EC / 16 / 8;   // 3125
    for (int l = 0; l < NLAYERS; ++l) {
        // s2d: gather y0 via src, scatter silu(out) into d1 at dst
        k_tp<<<tpBlocks, 256, 0, stream>>>(ef, y, src, dst,
                                           wfrag + (size_t)(l * 2 + 0) * FRAG_HALVES,
                                           dacc + (size_t)N0C * DIMC);
        // d2s: gather y1 via dst, scatter into d0 at src
        k_tp<<<tpBlocks, 256, 0, stream>>>(ef, y + (size_t)N0C * DIMC, dst, src,
                                           wfrag + (size_t)(l * 2 + 1) * FRAG_HALVES,
                                           dacc);
        k_addzero<<<(int)((NY + 255) / 256), 256, 0, stream>>>(y, dacc, (int)NY);
    }

    k_out_init<<<1, 1, 0, stream>>>(out);
    k_readout<<<(N0C + N1C + 255) / 256, 256, 0, stream>>>(y, ro_w, ro_b, out);
}
